// AttentionLayer_82884278879004
// MI455X (gfx1250) — matrix operand order repaired and checked by reading, not compiled, not run on hardware
//
#include <hip/hip_runtime.h>
#include <hip/hip_bf16.h>

// ---------------------------------------------------------------------------
// CDNA5 (gfx1250) fused attention layer, bf16 WMMA everywhere:
//   K0: W_in, W_o  -> bf16 transposed weight copies (B-fragment friendly)
//   K1: QKV = x0 @ W_in           -> bf16 Q[bh,s,64], K[bh,s,64], Vt[bh,64,s]
//   K2: flash attention (wave32)  -> fp32 O[b,s,d]
//   K3: y = O @ W_o + x0          -> d_out (fp32)
//   K4: LayerNorm in-place on d_out
// ---------------------------------------------------------------------------

typedef __bf16 v16bf __attribute__((ext_vector_type(16)));
typedef __bf16 v8bf  __attribute__((ext_vector_type(8)));
typedef __bf16 v4bf  __attribute__((ext_vector_type(4)));
typedef float  v8f   __attribute__((ext_vector_type(8)));
typedef float  v4f   __attribute__((ext_vector_type(4)));

#define WMMA_BF16(A, B, C) \
  __builtin_amdgcn_wmma_f32_16x16x32_bf16(false, (A), false, (B), (short)0, (C), false, false)

#define NEG_BIG (-3.0e38f)

// ---------------------------------------------------------------------------
// Kernel 0: weight convert + transpose.  W: [K=1024, N] f32 row-major ->
// WT: [N, 1024] bf16 row-major (so a WMMA B-fragment is one contiguous v16bf).
// One block per output row n; 256 threads x 4 k-elements.
// ---------------------------------------------------------------------------
__global__ __launch_bounds__(256) void wcvt_kernel(
    const float* __restrict__ W, unsigned short* __restrict__ wt, int N)
{
  __bf16* WT = reinterpret_cast<__bf16*>(wt);
  const int n  = blockIdx.x;
  const int k0 = threadIdx.x * 4;
  v4bf o;
#pragma unroll
  for (int i = 0; i < 4; ++i) o[i] = (__bf16)W[(size_t)(k0 + i) * N + n];
  *(v4bf*)(WT + (size_t)n * 1024 + k0) = o;
}

// ---------------------------------------------------------------------------
// Kernel 1: QKV projection. One wave -> 32(M) x 64(N) tile; 8 waves/block.
// X: [8192,1024] f32.  WT: [3072,1024] bf16 (transposed W_in).
// ---------------------------------------------------------------------------
__global__ __launch_bounds__(256) void qkv_gemm_kernel(
    const float* __restrict__ X, const unsigned short* __restrict__ wt,
    unsigned short* __restrict__ qw, unsigned short* __restrict__ kw,
    unsigned short* __restrict__ vw)
{
  const __bf16* WT = reinterpret_cast<const __bf16*>(wt);
  __bf16* Qb = reinterpret_cast<__bf16*>(qw);
  __bf16* Kb = reinterpret_cast<__bf16*>(kw);
  __bf16* Vt = reinterpret_cast<__bf16*>(vw);

  const int lane = threadIdx.x & 31;
  const int wave = threadIdx.x >> 5;
  const int m0   = (blockIdx.y * 8 + wave) * 32;  // 0..8160
  const int n0   = blockIdx.x * 64;               // 0..3071
  const int row  = lane & 15;
  const int grp  = lane >> 4;
  const int kb8  = grp * 8;     // unused: A-frag K base is khalf too
  const int khalf = grp * 16;   // B-frag K base

  v8f acc[2][4] = {};
  for (int k0 = 0; k0 < 1024; k0 += 32) {
    v16bf af[2];
#pragma unroll
    for (int u = 0; u < 2; ++u) {
      const float* xrow = X + (size_t)(m0 + u * 16 + row) * 1024 + k0;
      __builtin_prefetch(xrow + 32, 0, 1);        // -> global_prefetch_b8
      v4f a0 = *(const v4f*)(xrow + khalf);
      v4f a1 = *(const v4f*)(xrow + khalf + 4);
      v4f a2 = *(const v4f*)(xrow + khalf + 8);
      v4f a3 = *(const v4f*)(xrow + khalf + 12);
#pragma unroll
      for (int i = 0; i < 4; ++i) {
        af[u][i]      = (__bf16)a0[i];
        af[u][4 + i]  = (__bf16)a1[i];
        af[u][8 + i]  = (__bf16)a2[i];
        af[u][12 + i] = (__bf16)a3[i];
      }
    }
#pragma unroll
    for (int t = 0; t < 4; ++t) {
      const int n = n0 + t * 16 + row;            // B-frag: N = lane&15
      v16bf bfrag = *(const v16bf*)(WT + (size_t)n * 1024 + k0 + khalf);
      acc[0][t] = WMMA_BF16(af[0], bfrag, acc[0][t]);
      acc[1][t] = WMMA_BF16(af[1], bfrag, acc[1][t]);
    }
  }

  // Scatter into head-interleaved q/k/v layout (192 cols per head: q|k|v)
#pragma unroll
  for (int t = 0; t < 4; ++t) {
    const int n    = n0 + t * 16 + row;
    const int h    = n / 192;
    const int rem  = n % 192;
    const int kind = rem / 64;
    const int dc   = rem % 64;
#pragma unroll
    for (int u = 0; u < 2; ++u) {
#pragma unroll
      for (int r = 0; r < 8; ++r) {
        const int m    = m0 + u * 16 + r + 8 * grp;  // C-frag: M = r + 8*grp
        const int bidx = m >> 11;
        const int srow = m & 2047;
        const size_t bh = (size_t)bidx * 16 + h;
        const __bf16 v = (__bf16)acc[u][t][r];
        if (kind == 0)      Qb[(bh * 2048 + srow) * 64 + dc] = v;
        else if (kind == 1) Kb[(bh * 2048 + srow) * 64 + dc] = v;
        else                Vt[(bh * 64 + dc) * 2048 + srow] = v;
      }
    }
  }
}

// ---------------------------------------------------------------------------
// Kernel 2: causal flash attention. One wave per (bh, 16-query-row tile).
// ---------------------------------------------------------------------------
__global__ __launch_bounds__(256) void attn_kernel(
    const unsigned short* __restrict__ qw, const unsigned short* __restrict__ kw,
    const unsigned short* __restrict__ vw, float* __restrict__ O)
{
  __shared__ __align__(16) __bf16 plds[8][16 * 32];  // per-wave 16x32 P tile

  const __bf16* Qb = reinterpret_cast<const __bf16*>(qw);
  const __bf16* Kb = reinterpret_cast<const __bf16*>(kw);
  const __bf16* Vt = reinterpret_cast<const __bf16*>(vw);

  const int lane = threadIdx.x & 31;
  const int wave = threadIdx.x >> 5;
  const int bh   = blockIdx.y;                    // 0..63 (b*16 + h)
  const int q0   = (blockIdx.x * 8 + wave) * 16;  // query tile start
  const int row  = lane & 15;
  const int grp  = lane >> 4;
  const int kb8  = grp * 8;
  const int khalf = grp * 16;

  const __bf16* Qbh = Qb + (size_t)bh * 2048 * 64;
  const __bf16* Kbh = Kb + (size_t)bh * 2048 * 64;
  const __bf16* Vbh = Vt + (size_t)bh * 64 * 2048;

  // Q A-fragments for dh=64 (two K=32 slices), loaded once.
  v16bf qf[2];
#pragma unroll
  for (int kf = 0; kf < 2; ++kf) {
    const __bf16* qrow = Qbh + (size_t)(q0 + row) * 64 + kf * 32;
    v8bf lo = *(const v8bf*)(qrow + khalf);
    v8bf hi = *(const v8bf*)(qrow + khalf + 8);
#pragma unroll
    for (int i = 0; i < 8; ++i) { qf[kf][i] = lo[i]; qf[kf][8 + i] = hi[i]; }
  }

  v8f o[4] = {};
  float mrow[8], lrow[8];
#pragma unroll
  for (int r = 0; r < 8; ++r) { mrow[r] = NEG_BIG; lrow[r] = 0.0f; }

  const float scale = 0.125f;              // 1/sqrt(64)
  const int kbmax = (q0 + 15) / 32;        // causal bound on 32-wide key blocks

  for (int kb = 0; kb <= kbmax; ++kb) {
    const int j0 = kb * 32;

    // ---- S = Q K^T for a 16x32 key block (two 16x16 tiles) ----
    v8f s[2] = {};
#pragma unroll
    for (int nt = 0; nt < 2; ++nt) {
      const int j = j0 + nt * 16 + row;    // key index carried by lane (B: N=lane)
#pragma unroll
      for (int kf = 0; kf < 2; ++kf) {
        v16bf bfrag = *(const v16bf*)(Kbh + (size_t)j * 64 + kf * 32 + khalf);
        s[nt] = WMMA_BF16(qf[kf], bfrag, s[nt]);
      }
    }

    // ---- scale + causal mask + fp32 online softmax ----
#pragma unroll
    for (int nt = 0; nt < 2; ++nt) {
      const int j = j0 + nt * 16 + row;
#pragma unroll
      for (int r = 0; r < 8; ++r) {
        const int m = q0 + r + 8 * grp;
        float v = s[nt][r] * scale;
        s[nt][r] = (j > m) ? NEG_BIG : v;
      }
    }
#pragma unroll
    for (int r = 0; r < 8; ++r) {
      float rm = fmaxf(s[0][r], s[1][r]);
      rm = fmaxf(rm, __shfl_xor(rm, 1, 32));
      rm = fmaxf(rm, __shfl_xor(rm, 2, 32));
      rm = fmaxf(rm, __shfl_xor(rm, 4, 32));
      rm = fmaxf(rm, __shfl_xor(rm, 8, 32));
      const float mn = fmaxf(mrow[r], rm);
      const float al = __expf(mrow[r] - mn);
      mrow[r] = mn;
      const float p0 = __expf(s[0][r] - mn);
      const float p1 = __expf(s[1][r] - mn);
      s[0][r] = p0; s[1][r] = p1;
      float rs = p0 + p1;
      rs += __shfl_xor(rs, 1, 32);
      rs += __shfl_xor(rs, 2, 32);
      rs += __shfl_xor(rs, 4, 32);
      rs += __shfl_xor(rs, 8, 32);
      lrow[r] = lrow[r] * al + rs;
      o[0][r] *= al; o[1][r] *= al; o[2][r] *= al; o[3][r] *= al;
    }

    // ---- C-fragment -> A-fragment layout swap for P via per-wave LDS ----
    __bf16* pw = &plds[wave][0];
#pragma unroll
    for (int nt = 0; nt < 2; ++nt)
#pragma unroll
      for (int r = 0; r < 8; ++r)
        pw[(r + 8 * grp) * 32 + nt * 16 + row] = (__bf16)s[nt][r];
    asm volatile("s_wait_dscnt 0x0" ::: "memory");  // LDS stores visible to this wave
    v16bf pf;
    {
      const __bf16* pr = pw + row * 32;
      v8bf lo = *(const v8bf*)(pr + khalf);
      v8bf hi = *(const v8bf*)(pr + khalf + 8);
#pragma unroll
      for (int i = 0; i < 8; ++i) { pf[i] = lo[i]; pf[8 + i] = hi[i]; }
    }

    // ---- O += P @ V (V transposed in ws: contiguous b128 B-fragments) ----
#pragma unroll
    for (int nt = 0; nt < 4; ++nt) {
      const int n = nt * 16 + row;         // dh column carried by lane
      v16bf vfrag = *(const v16bf*)(Vbh + (size_t)n * 2048 + j0 + khalf);
      o[nt] = WMMA_BF16(pf, vfrag, o[nt]);
    }
  }

  // ---- normalize + write O[b, s, d] fp32 ----
  const int b = bh >> 4;
  const int h = bh & 15;
#pragma unroll
  for (int nt = 0; nt < 4; ++nt) {
    const int n = nt * 16 + row;
#pragma unroll
    for (int r = 0; r < 8; ++r) {
      const int m = q0 + r + 8 * grp;
      O[((size_t)b * 2048 + m) * 1024 + h * 64 + n] = o[nt][r] / lrow[r];
    }
  }
}

// ---------------------------------------------------------------------------
// Kernel 3: y = O @ W_o + x0 -> d_out. One wave -> 32x64 tile; WT = bf16 W_o^T.
// ---------------------------------------------------------------------------
__global__ __launch_bounds__(256) void out_proj_kernel(
    const float* __restrict__ O, const unsigned short* __restrict__ wt,
    const float* __restrict__ X0, float* __restrict__ Y)
{
  const __bf16* WT = reinterpret_cast<const __bf16*>(wt);
  const int lane = threadIdx.x & 31;
  const int wave = threadIdx.x >> 5;
  const int m0   = (blockIdx.y * 8 + wave) * 32;
  const int n0   = blockIdx.x * 64;
  const int row  = lane & 15;
  const int grp  = lane >> 4;
  const int kb8  = grp * 8;
  const int khalf = grp * 16;

  v8f acc[2][4] = {};
  for (int k0 = 0; k0 < 1024; k0 += 32) {
    v16bf af[2];
#pragma unroll
    for (int u = 0; u < 2; ++u) {
      const float* orow = O + (size_t)(m0 + u * 16 + row) * 1024 + k0;
      __builtin_prefetch(orow + 32, 0, 1);
      v4f a0 = *(const v4f*)(orow + khalf);
      v4f a1 = *(const v4f*)(orow + khalf + 4);
      v4f a2 = *(const v4f*)(orow + khalf + 8);
      v4f a3 = *(const v4f*)(orow + khalf + 12);
#pragma unroll
      for (int i = 0; i < 4; ++i) {
        af[u][i]      = (__bf16)a0[i];
        af[u][4 + i]  = (__bf16)a1[i];
        af[u][8 + i]  = (__bf16)a2[i];
        af[u][12 + i] = (__bf16)a3[i];
      }
    }
#pragma unroll
    for (int t = 0; t < 4; ++t) {
      const int n = n0 + t * 16 + row;
      v16bf bfrag = *(const v16bf*)(WT + (size_t)n * 1024 + k0 + khalf);
      acc[0][t] = WMMA_BF16(af[0], bfrag, acc[0][t]);
      acc[1][t] = WMMA_BF16(af[1], bfrag, acc[1][t]);
    }
  }
#pragma unroll
  for (int t = 0; t < 4; ++t) {
    const int n = n0 + t * 16 + row;
#pragma unroll
    for (int u = 0; u < 2; ++u) {
#pragma unroll
      for (int r = 0; r < 8; ++r) {
        const size_t m = m0 + u * 16 + r + 8 * grp;
        Y[m * 1024 + n] = acc[u][t][r] + X0[m * 1024 + n];
      }
    }
  }
}

// ---------------------------------------------------------------------------
// Kernel 4: LayerNorm (no affine) in-place. One 256-thread block per row.
// All reads happen before __syncthreads(); all writes after -> in-place safe.
// ---------------------------------------------------------------------------
__global__ __launch_bounds__(256) void layernorm_kernel(float* __restrict__ Y)
{
  __shared__ float red_s[8];
  __shared__ float red_ss[8];
  const size_t rowi = blockIdx.x;
  float* y = Y + rowi * 1024;

  v4f vals = *(const v4f*)(y + threadIdx.x * 4);
  float s = 0.0f, ss = 0.0f;
#pragma unroll
  for (int i = 0; i < 4; ++i) { s += vals[i]; ss += vals[i] * vals[i]; }
#pragma unroll
  for (int off = 16; off >= 1; off >>= 1) {
    s  += __shfl_xor(s, off, 32);
    ss += __shfl_xor(ss, off, 32);
  }
  const int lane = threadIdx.x & 31;
  const int wave = threadIdx.x >> 5;
  if (lane == 0) { red_s[wave] = s; red_ss[wave] = ss; }
  __syncthreads();
  float ts = 0.0f, tss = 0.0f;
#pragma unroll
  for (int i = 0; i < 8; ++i) { ts += red_s[i]; tss += red_ss[i]; }
  const float mu  = ts * (1.0f / 1024.0f);
  const float var = tss * (1.0f / 1024.0f) - mu * mu;
  const float inv = rsqrtf(var + 1e-5f);
  v4f outv;
#pragma unroll
  for (int i = 0; i < 4; ++i) outv[i] = (vals[i] - mu) * inv;
  *(v4f*)(y + threadIdx.x * 4) = outv;
}

// ---------------------------------------------------------------------------
extern "C" void kernel_launch(void* const* d_in, const int* in_sizes, int n_in,
                              void* d_out, int out_size, void* d_ws, size_t ws_size,
                              hipStream_t stream)
{
  (void)in_sizes; (void)n_in; (void)out_size; (void)ws_size;
  const float* x0  = (const float*)d_in[0];   // [4,2048,1024]
  const float* Win = (const float*)d_in[1];   // [1024,3072]
  const float* Wo  = (const float*)d_in[2];   // [1024,1024]
  // d_in[3] = src_mask, all-false per spec -> unused
  float* out = (float*)d_out;                 // [4,2048,1024] fp32

  char* ws = (char*)d_ws;
  const size_t MB = (size_t)1024 * 1024;
  unsigned short* Qb   = (unsigned short*)(ws);             // 16 MB bf16
  unsigned short* Kb   = (unsigned short*)(ws + 16 * MB);   // 16 MB bf16
  unsigned short* Vt   = (unsigned short*)(ws + 32 * MB);   // 16 MB bf16
  float*          O    = (float*)         (ws + 48 * MB);   // 32 MB f32
  unsigned short* WinT = (unsigned short*)(ws + 80 * MB);   //  6 MB bf16 [3072,1024]
  unsigned short* WoT  = (unsigned short*)(ws + 88 * MB);   //  2 MB bf16 [1024,1024]

  // K0: transpose + bf16-convert weights (B-fragments become contiguous b128s)
  wcvt_kernel<<<dim3(3072), 256, 0, stream>>>(Win, WinT, 3072);
  wcvt_kernel<<<dim3(1024), 256, 0, stream>>>(Wo, WoT, 1024);

  // K1: QKV projection. Grid: N-tiles(3072/64=48) x M-blocks(8192/256=32)
  qkv_gemm_kernel<<<dim3(48, 32), 256, 0, stream>>>(x0, WinT, Qb, Kb, Vt);

  // K2: attention. Grid: q-tile-blocks(128/8=16) x bh(64)
  attn_kernel<<<dim3(16, 64), 256, 0, stream>>>(Qb, Kb, Vt, O);

  // K3: out projection + residual, straight into d_out
  out_proj_kernel<<<dim3(16, 32), 256, 0, stream>>>(O, WoT, x0, out);

  // K4: LayerNorm in-place on d_out
  layernorm_kernel<<<dim3(8192), 256, 0, stream>>>(out);
}
